// SE3AllFeatures_87548613361691
// MI455X (gfx1250) — matrix operand order repaired and checked by reading, not compiled, not run on hardware
//
#include <hip/hip_runtime.h>
#include <math.h>

typedef __attribute__((ext_vector_type(16))) _Float16 v16h;
typedef __attribute__((ext_vector_type(8)))  _Float16 v8h;
typedef __attribute__((ext_vector_type(8)))  float    v8f;

#define NNODES 50000
#define NEDGES 200000
#define NGRAPHS 64
#define HDIM 60
#define WNUMEL 864
#define NHID 64

// ---- path tables (PATHS order from the reference) ----
constexpr int P_L1[11]   = {0,0,0,1,1,1,1,2,2,2,2};
constexpr int P_L2[11]   = {0,1,2,0,1,1,2,0,1,2,2};
constexpr int P_LO[11]   = {0,1,2,1,0,2,1,2,1,0,2};
constexpr int P_WOFF[11] = {0,256,384,448,512,640,672,736,752,784,848};
constexpr int P_COFF[11] = {0,1,10,35,44,53,98,143,168,213,238};
constexpr int HIDL[3] = {16,8,4};
constexpr int XOFF[3] = {0,16,40};
constexpr int SHOFF[3] = {0,1,4};

__device__ __forceinline__ float siluf(float x){ return x / (1.f + __expf(-x)); }

// ============================ Wigner 3j init ============================
__device__ double dfact(int n){ double r=1.0; for(int i=2;i<=n;++i) r*=(double)i; return r; }

__device__ double su2cg(int j1,int j2,int j3,int m1,int m2,int m3){
  if (m3 != m1+m2) return 0.0;
  int vmin = -j1+j2+m3; if (-j1+m1>vmin) vmin=-j1+m1; if (0>vmin) vmin=0;
  int vmax = j2+j3+m1; if (j3-j1+j2<vmax) vmax=j3-j1+j2; if (j3+m3<vmax) vmax=j3+m3;
  double c = sqrt(((double)(2*j3+1))*dfact(j3+j1-j2)*dfact(j3-j1+j2)*dfact(j1+j2-j3)*
                  dfact(j3+m3)*dfact(j3-m3) /
                  (dfact(j1+j2+j3+1)*dfact(j1-m1)*dfact(j1+m1)*dfact(j2-m2)*dfact(j2+m2)));
  double s = 0.0;
  for (int v=vmin; v<=vmax; ++v){
    double sg = ((v+j2+m2)&1) ? -1.0 : 1.0;
    s += sg*dfact(j2+j3+m1-v)*dfact(j1-m1+v) /
         (dfact(v)*dfact(j3-j1+j2-v)*dfact(j3+m3-v)*dfact(v+j1-j2-m3));
  }
  return c*s;
}

struct cpx { double re, im; };
__device__ __forceinline__ cpx cmul(cpx a, cpx b){
  cpx r; r.re=a.re*b.re-a.im*b.im; r.im=a.re*b.im+a.im*b.re; return r;
}

__device__ void buildq(int l, cpx q[5][5]){
  for(int a=0;a<5;++a) for(int b=0;b<5;++b){ q[a][b].re=0.0; q[a][b].im=0.0; }
  double is = 0.70710678118654752440;
  for(int m=-l;m<0;++m){ q[l+m][l-m].re = is; q[l+m][l+m].im = -is; }
  q[l][l].re = 1.0;
  for(int m=1;m<=l;++m){
    double sg = (m&1) ? -1.0 : 1.0;
    q[l+m][l+m].re = sg*is;
    q[l+m][l-m].im = sg*is;
  }
  cpx f;
  if (l==0){ f.re=1.0; f.im=0.0; }
  else if (l==1){ f.re=0.0; f.im=-1.0; }
  else { f.re=-1.0; f.im=0.0; }
  for(int a=0;a<2*l+1;++a) for(int b=0;b<2*l+1;++b) q[a][b]=cmul(f,q[a][b]);
}

__global__ void init_w3j_kernel(float* __restrict__ w3j){
  int p = threadIdx.x;
  if (p >= 11) return;
  int l1=P_L1[p], l2=P_L2[p], l3=P_LO[p];
  cpx q1[5][5], q2[5][5], q3[5][5];
  buildq(l1,q1); buildq(l2,q2); buildq(l3,q3);
  int d1=2*l1+1, d2=2*l2+1, d3=2*l3+1;
  double tmp[125]; double nrm=0.0;
  for(int jj=0;jj<d1;++jj) for(int ll=0;ll<d2;++ll) for(int mm=0;mm<d3;++mm){
    double ar = 0.0;
    for(int i=0;i<d1;++i) for(int k=0;k<d2;++k) for(int n=0;n<d3;++n){
      double cg = su2cg(l1,l2,l3, i-l1, k-l2, n-l3);
      if (cg == 0.0) continue;
      cpx t = cmul(q1[i][jj], q2[k][ll]);
      cpx c3; c3.re = q3[n][mm].re; c3.im = -q3[n][mm].im;
      t = cmul(t, c3);
      ar += t.re * cg;
    }
    tmp[(jj*d2+ll)*d3+mm] = ar;
    nrm += ar*ar;
  }
  double inv = (nrm > 0.0) ? 1.0/sqrt(nrm) : 0.0;
  float* out = w3j + P_COFF[p];
  for(int i=0;i<d1*d2*d3;++i) out[i] = (float)(tmp[i]*inv);
}

// ============================ misc kernels ============================
__global__ void fill_zero_kernel(float* __restrict__ p, int n){
  int i = blockIdx.x*blockDim.x + threadIdx.x;
  if (i < n) p[i] = 0.f;
}

__global__ void prep_w2t_kernel(const float* __restrict__ w2, _Float16* __restrict__ w2t){
  int idx = blockIdx.x*blockDim.x + threadIdx.x;
  if (idx >= WNUMEL*NHID) return;
  int col = idx >> 6, k = idx & 63;
  w2t[idx] = (_Float16)w2[k*WNUMEL + col];   // w2t[col][k]
}

// input embedding + input irreps_linear -> h (N,60)
__global__ void node_input_kernel(const float* __restrict__ xs, const int* __restrict__ ntype,
    const int* __restrict__ pmask, const int* __restrict__ lmask,
    const float* __restrict__ xv, const float* __restrict__ xt,
    const float* __restrict__ emb, const float* __restrict__ w0,
    const float* __restrict__ w1, const float* __restrict__ w2,
    float* __restrict__ h){
  int n = blockIdx.x*blockDim.x + threadIdx.x;
  if (n >= NNODES) return;
  float in0[20];
  #pragma unroll
  for(int i=0;i<10;++i) in0[i] = xs[n*10+i];
  int t = ntype[n];
  #pragma unroll
  for(int i=0;i<8;++i) in0[10+i] = emb[t*8+i];
  in0[18] = (float)pmask[n];
  in0[19] = (float)lmask[n];
  const float k0 = 0.22360679775f;   // 1/sqrt(20)
  const float k1 = 0.57735026919f;   // 1/sqrt(3)
  const float k2 = 0.70710678119f;   // 1/sqrt(2)
  float* hn = h + (long)n*HDIM;
  for(int w=0;w<16;++w){
    float s=0.f;
    #pragma unroll
    for(int u=0;u<20;++u) s += in0[u]*w0[u*16+w];
    hn[w] = s*k0;
  }
  for(int w=0;w<8;++w)
    #pragma unroll
    for(int i=0;i<3;++i){
      float s=0.f;
      #pragma unroll
      for(int u=0;u<3;++u) s += xv[n*9+u*3+i]*w1[u*8+w];
      hn[16+w*3+i] = s*k1;
    }
  for(int w=0;w<4;++w)
    #pragma unroll
    for(int i=0;i<5;++i){
      float s=0.f;
      #pragma unroll
      for(int u=0;u<2;++u) s += xt[n*10+u*5+i]*w2[u*4+w];
      hn[40+w*5+i] = s*k2;
    }
}

// edge MLP layer1: ef[10] -> silu -> hidden[64] (f16, WMMA A operand)
__global__ void edge_hidden_kernel(const float* __restrict__ radial, const float* __restrict__ cutoff,
    const float* __restrict__ lengths, const float* __restrict__ w1, const float* __restrict__ b1,
    _Float16* __restrict__ hidden){
  int idx = blockIdx.x*blockDim.x + threadIdx.x;
  if (idx >= NEDGES*NHID) return;
  int e = idx >> 6, j = idx & 63;
  float ef[10];
  #pragma unroll
  for(int i=0;i<8;++i) ef[i] = radial[e*8+i];
  ef[8] = cutoff[e]; ef[9] = lengths[e];
  float s = b1[j];
  #pragma unroll
  for(int i=0;i<10;++i) s += ef[i]*w1[i*64+j];
  hidden[idx] = (_Float16)siluf(s);
}

// ============================ edge message (WMMA + tensor product) ============================
__launch_bounds__(128)
__global__ void edge_msg_kernel(const _Float16* __restrict__ hidden, const _Float16* __restrict__ w2t,
    const float* __restrict__ b2, const float* __restrict__ h, const int* __restrict__ eidx,
    const float* __restrict__ sh0, const float* __restrict__ sh1, const float* __restrict__ sh2,
    const float* __restrict__ w3j_g, float* __restrict__ agg){
  __shared__ float w_lds[16][WNUMEL+4];   // per-edge weights w = hidden@W2 + b2
  __shared__ float xs_lds[16][HDIM];      // h[src] per edge
  __shared__ float sh_lds[16][9];         // spherical harmonics per edge
  __shared__ float w3j_s[384];

  const int tid = threadIdx.x;
  const int e0  = blockIdx.x * 16;

  for(int i=tid;i<363;i+=128) w3j_s[i] = w3j_g[i];
  for(int i=tid;i<16*HDIM;i+=128){
    int m = i/HDIM, d = i%HDIM;
    int e = e0 + m; if (e >= NEDGES) e = NEDGES-1;
    int src = eidx[e];
    xs_lds[m][d] = h[(long)src*HDIM + d];
  }
  for(int i=tid;i<16*9;i+=128){
    int m = i/9, d = i%9;
    int e = e0 + m; if (e >= NEDGES) e = NEDGES-1;
    float v;
    if (d==0)      v = sh0[e];
    else if (d<4)  v = sh1[e*3 + (d-1)];
    else           v = sh2[e*5 + (d-4)];
    sh_lds[m][d] = v;
  }

  // ---------------- phase 1: w[16 edges][864] = hidden @ W2 + b2, via WMMA ----------------
  const int lane = tid & 31;
  const int wv   = tid >> 5;         // wave id 0..3
  const int row  = lane & 15;        // A: M row / B,D: N column
  const int sel  = lane >> 4;        // lane half

  // A (16x64 f16): interleaved K layout: half h -> K = (h/8)*16 + sel*8 + (h%8)
  const _Float16* hp = hidden + (long)(e0 + row)*NHID;
  v8h a00 = *(const v8h*)(hp +      sel*8);
  v8h a01 = *(const v8h*)(hp + 16 + sel*8);
  v8h a10 = *(const v8h*)(hp + 32 + sel*8);
  v8h a11 = *(const v8h*)(hp + 48 + sel*8);
  v16h A0 = __builtin_shufflevector(a00,a01,0,1,2,3,4,5,6,7,8,9,10,11,12,13,14,15);
  v16h A1 = __builtin_shufflevector(a10,a11,0,1,2,3,4,5,6,7,8,9,10,11,12,13,14,15);

  for (int t = wv; t < 54; t += 4){
    const int col0 = t*16;
    // B (64x16 f16) from W2T[col][k]: half h -> K = (h/8)*16 + sel*8 + (h%8) (+32 for second WMMA)
    const _Float16* bp = w2t + (long)(col0 + row)*NHID;
    v8h b00 = *(const v8h*)(bp +      sel*8);
    v8h b01 = *(const v8h*)(bp + 16 + sel*8);
    v8h b10 = *(const v8h*)(bp + 32 + sel*8);
    v8h b11 = *(const v8h*)(bp + 48 + sel*8);
    v16h B0 = __builtin_shufflevector(b00,b01,0,1,2,3,4,5,6,7,8,9,10,11,12,13,14,15);
    v16h B1 = __builtin_shufflevector(b10,b11,0,1,2,3,4,5,6,7,8,9,10,11,12,13,14,15);
    v8f acc = {0.f,0.f,0.f,0.f,0.f,0.f,0.f,0.f};
    acc = __builtin_amdgcn_wmma_f32_16x16x32_f16(false, A0, false, B0, (short)0, acc, false, false);
    acc = __builtin_amdgcn_wmma_f32_16x16x32_f16(false, A1, false, B1, (short)0, acc, false, false);
    const float bias = b2[col0 + row];
    #pragma unroll
    for(int r=0;r<8;++r)
      w_lds[r + 8*sel][col0 + row] = acc[r] + bias;
  }
  __syncthreads();

  // ---------------- phase 2: tensor product + scatter ----------------
  const int m   = tid & 15;        // edge in tile
  const int grp = tid >> 4;        // 8 threads per edge
  const int e   = e0 + m;
  const int dst = (e < NEDGES) ? eidx[NEDGES + e] : 0;
  const float nrm0 = 0.18898223650f;  // sqrt(1/28)
  const float nrm1 = 0.28867513459f;  // sqrt(3/36)
  const float nrm2 = 0.39528470752f;  // sqrt(5/32)

  int send = grp*8 + 8; if (send > 60) send = 60;
  for (int s = grp*8; s < send; ++s){
    int lo, wch, k;
    if (s < 16){ lo=0; wch=s;           k=0; }
    else if (s < 40){ lo=1; wch=(s-16)/3; k=(s-16)%3; }
    else { lo=2; wch=(s-40)/5; k=(s-40)%5; }
    float accv = 0.f;
    #pragma unroll
    for (int p=0;p<11;++p){
      if (P_LO[p] != lo) continue;
      const int l1 = P_L1[p], l2 = P_L2[p];
      const int d1 = 2*l1+1, d2 = 2*l2+1, d3 = 2*lo+1;
      const float* C  = w3j_s + P_COFF[p];
      const float* sv = &sh_lds[m][SHOFF[l2]];
      float G[5];
      for(int i=0;i<d1;++i){
        float g = 0.f;
        for(int j=0;j<d2;++j) g += C[(i*d2+j)*d3 + k] * sv[j];
        G[i] = g;
      }
      const int U  = HIDL[l1];
      const int Wd = HIDL[lo];
      const float* xv = &xs_lds[m][XOFF[l1]];
      const float* Wp = &w_lds[m][P_WOFF[p]];
      float val = 0.f;
      for(int u=0;u<U;++u){
        float t = 0.f;
        for(int i=0;i<d1;++i) t += xv[u*d1+i]*G[i];
        val += Wp[u*Wd + wch]*t;
      }
      accv += val;
    }
    accv *= (lo==0) ? nrm0 : ((lo==1) ? nrm1 : nrm2);
    if (e < NEDGES) atomicAdd(&agg[(long)dst*HDIM + s], accv);
  }
}

// skip irreps_linear + agg, then silu + layernorm
__global__ void node_update_kernel(const float* __restrict__ h, const float* __restrict__ agg,
    const float* __restrict__ sw0, const float* __restrict__ sw1, const float* __restrict__ sw2,
    const float* __restrict__ ln_g, const float* __restrict__ ln_b, float* __restrict__ hout){
  int n = blockIdx.x*blockDim.x + threadIdx.x;
  if (n >= NNODES) return;
  float hv[60], o[60];
  const float* hn = h + (long)n*HDIM;
  #pragma unroll
  for(int d=0;d<60;++d) hv[d] = hn[d];
  const float i4 = 0.25f, i8 = 0.35355339059f, i2 = 0.5f;
  for(int w=0;w<16;++w){
    float s=0.f;
    #pragma unroll
    for(int u=0;u<16;++u) s += hv[u]*sw0[u*16+w];
    o[w] = s*i4;
  }
  for(int w=0;w<8;++w)
    #pragma unroll
    for(int i=0;i<3;++i){
      float s=0.f;
      #pragma unroll
      for(int u=0;u<8;++u) s += hv[16+u*3+i]*sw1[u*8+w];
      o[16+w*3+i] = s*i8;
    }
  for(int w=0;w<4;++w)
    #pragma unroll
    for(int i=0;i<5;++i){
      float s=0.f;
      #pragma unroll
      for(int u=0;u<4;++u) s += hv[40+u*5+i]*sw2[u*4+w];
      o[40+w*5+i] = s*i2;
    }
  const float* an = agg + (long)n*HDIM;
  float x[60]; float mu = 0.f;
  #pragma unroll
  for(int d=0;d<60;++d){
    float t = siluf(o[d] + an[d]);
    x[d] = t; mu += t;
  }
  mu *= (1.f/60.f);
  float var = 0.f;
  #pragma unroll
  for(int d=0;d<60;++d){ float dd = x[d]-mu; var += dd*dd; }
  var *= (1.f/60.f);
  float inv = rsqrtf(var + 1e-5f);
  float* on = hout + (long)n*HDIM;
  #pragma unroll
  for(int d=0;d<60;++d) on[d] = (x[d]-mu)*inv*ln_g[d] + ln_b[d];
}

// readout linear + masked segment sums (atomics per graph)
__global__ void readout_kernel(const float* __restrict__ h, const int* __restrict__ batch,
    const int* __restrict__ pmask, const int* __restrict__ lmask,
    const float* __restrict__ ro, float* __restrict__ gsums){
  int n = blockIdx.x*blockDim.x + threadIdx.x;
  if (n >= NNODES) return;
  int g = batch[n];
  float pm = (float)pmask[n], lm = (float)lmask[n];
  float* prot = gsums;
  float* lig  = gsums + NGRAPHS*32;
  float* alls = gsums + 2*NGRAPHS*32;
  float* cnt  = gsums + 3*NGRAPHS*32;
  const float* hn = h + (long)n*HDIM;
  for(int w=0;w<32;++w){
    float s = 0.f;
    #pragma unroll
    for(int u=0;u<16;++u) s += hn[u]*ro[u*32+w];
    s *= 0.25f;   // 1/sqrt(16)
    if (pm != 0.f) atomicAdd(&prot[g*32+w], s*pm);
    if (lm != 0.f) atomicAdd(&lig[g*32+w], s*lm);
    atomicAdd(&alls[g*32+w], s);
  }
  if (pm != 0.f) atomicAdd(&cnt[g], pm);
  if (lm != 0.f) atomicAdd(&cnt[NGRAPHS+g], lm);
  atomicAdd(&cnt[2*NGRAPHS+g], 1.f);
}

__global__ void head_kernel(const float* __restrict__ gsums, const float* __restrict__ w1,
    const float* __restrict__ b1, const float* __restrict__ w2, const float* __restrict__ b2,
    float* __restrict__ out){
  int g = threadIdx.x;
  if (g >= NGRAPHS) return;
  const float* prot = gsums;
  const float* lig  = gsums + NGRAPHS*32;
  const float* alls = gsums + 2*NGRAPHS*32;
  const float* cnt  = gsums + 3*NGRAPHS*32;
  float cp = fmaxf(cnt[g],1.f), cl = fmaxf(cnt[NGRAPHS+g],1.f), ca = fmaxf(cnt[2*NGRAPHS+g],1.f);
  float gf[96];
  for(int w=0;w<32;++w){
    gf[w]    = prot[g*32+w]/cp;
    gf[32+w] = lig[g*32+w]/cl;
    gf[64+w] = alls[g*32+w]/ca;
  }
  float o = b2[0];
  for(int j=0;j<64;++j){
    float s = b1[j];
    for(int i=0;i<96;++i) s += gf[i]*w1[i*64+j];
    o += siluf(s)*w2[j];
  }
  out[g] = o;
}

// ============================ launcher ============================
extern "C" void kernel_launch(void* const* d_in, const int* in_sizes, int n_in,
                              void* d_out, int out_size, void* d_ws, size_t ws_size,
                              hipStream_t stream){
  (void)in_sizes; (void)n_in; (void)out_size; (void)ws_size;
  const float* x_scalar = (const float*)d_in[0];
  const int*   node_type= (const int*)d_in[1];
  const int*   pmask    = (const int*)d_in[2];
  const int*   lmask    = (const int*)d_in[3];
  const float* x_vector = (const float*)d_in[4];
  const float* x_tensor = (const float*)d_in[5];
  const int*   edge_idx = (const int*)d_in[6];
  const int*   batch    = (const int*)d_in[7];
  const float* e_radial = (const float*)d_in[8];
  const float* e_cutoff = (const float*)d_in[9];
  const float* e_len    = (const float*)d_in[10];
  const float* sh0      = (const float*)d_in[11];
  const float* sh1      = (const float*)d_in[12];
  const float* sh2      = (const float*)d_in[13];
  const float* emb      = (const float*)d_in[14];
  const float* in_w0    = (const float*)d_in[15];
  const float* in_w1    = (const float*)d_in[16];
  const float* in_w2    = (const float*)d_in[17];
  // mp1: 18..24  mp2: 25..31
  const float* ln_g     = (const float*)d_in[32];
  const float* ln_b     = (const float*)d_in[33];
  const float* ro_w0    = (const float*)d_in[34];
  const float* hd_w1    = (const float*)d_in[35];
  const float* hd_b1    = (const float*)d_in[36];
  const float* hd_w2    = (const float*)d_in[37];
  const float* hd_b2    = (const float*)d_in[38];

  // workspace layout (bytes), all offsets 256B-aligned; total ~61.9 MB
  char* ws = (char*)d_ws;
  float*    w3j    = (float*)(ws + 0);
  float*    hA     = (float*)(ws + 2048);
  float*    hB     = (float*)(ws + 2048 + 12000000);
  float*    agg    = (float*)(ws + 2048 + 24000000);
  _Float16* hidden = (_Float16*)(ws + 2048 + 36000000);
  _Float16* w2t1   = (_Float16*)(ws + 2048 + 36000000 + 25600000);
  _Float16* w2t2   = (_Float16*)(ws + 2048 + 36000000 + 25600000 + 110592);
  float*    gsums  = (float*)(ws + 2048 + 36000000 + 25600000 + 2*110592);

  init_w3j_kernel<<<1, 32, 0, stream>>>(w3j);
  prep_w2t_kernel<<<(WNUMEL*NHID+255)/256, 256, 0, stream>>>((const float*)d_in[20], w2t1);
  prep_w2t_kernel<<<(WNUMEL*NHID+255)/256, 256, 0, stream>>>((const float*)d_in[27], w2t2);
  node_input_kernel<<<(NNODES+255)/256, 256, 0, stream>>>(
      x_scalar, node_type, pmask, lmask, x_vector, x_tensor, emb, in_w0, in_w1, in_w2, hA);

  float* hcur = hA; float* hnext = hB;
  for (int L = 0; L < 2; ++L){
    int base = 18 + L*7;
    const float* mw1 = (const float*)d_in[base+0];
    const float* mb1 = (const float*)d_in[base+1];
    const float* mb2 = (const float*)d_in[base+3];
    const float* sk0 = (const float*)d_in[base+4];
    const float* sk1 = (const float*)d_in[base+5];
    const float* sk2 = (const float*)d_in[base+6];
    _Float16* w2t = (L == 0) ? w2t1 : w2t2;

    edge_hidden_kernel<<<(NEDGES*NHID+255)/256, 256, 0, stream>>>(
        e_radial, e_cutoff, e_len, mw1, mb1, hidden);
    fill_zero_kernel<<<(NNODES*HDIM+255)/256, 256, 0, stream>>>(agg, NNODES*HDIM);
    edge_msg_kernel<<<NEDGES/16, 128, 0, stream>>>(
        hidden, w2t, mb2, hcur, edge_idx, sh0, sh1, sh2, w3j, agg);
    node_update_kernel<<<(NNODES+255)/256, 256, 0, stream>>>(
        hcur, agg, sk0, sk1, sk2, ln_g, ln_b, hnext);
    float* t = hcur; hcur = hnext; hnext = t;
  }

  fill_zero_kernel<<<(NGRAPHS*99+255)/256, 256, 0, stream>>>(gsums, NGRAPHS*99);
  readout_kernel<<<(NNODES+255)/256, 256, 0, stream>>>(hcur, batch, pmask, lmask, ro_w0, gsums);
  head_kernel<<<1, 64, 0, stream>>>(gsums, hd_w1, hd_b1, hd_w2, hd_b2, (float*)d_out);
}
